// FourrierKernelAttention_32074815766701
// MI455X (gfx1250) — matrix operand order repaired and checked by reading, not compiled, not run on hardware
//
#include <hip/hip_runtime.h>
#include <hip/hip_bf16.h>
#include <math.h>

// MI455X / gfx1250, wave32. bf16 WMMA for attention GEMMs, f32 WMMA for projections.
// Staging uses CDNA5 async global->LDS loads (ASYNCcnt) where available.

typedef __attribute__((ext_vector_type(16))) __bf16 v16bf;
typedef __attribute__((ext_vector_type(8)))  float  v8f;
typedef __attribute__((ext_vector_type(2)))  float  v2f;

#define N_B     2
#define L_SEQ   2048
#define H_N     8
#define D_HEAD  64
#define D_MODEL 512
#define F_DIM   128   // 2*d Fourier features (cos || sin)

#if __has_builtin(__builtin_amdgcn_global_load_async_to_lds_b128)
#define HAS_ASYNC_LDS 1
#endif

// 16-byte global -> LDS copy (async direct-to-LDS on CDNA5, sync fallback).
static __device__ __forceinline__ void copy16_g2l(const void* g, void* l) {
#ifdef HAS_ASYNC_LDS
  typedef int v4i __attribute__((vector_size(16)));
  typedef __attribute__((address_space(1))) v4i* g4p;
  typedef __attribute__((address_space(3))) v4i* l4p;
  __builtin_amdgcn_global_load_async_to_lds_b128((g4p)g, (l4p)l, 0, 0);
#else
  *(uint4*)l = *(const uint4*)g;
#endif
}

static __device__ __forceinline__ void async_wait() {
#ifdef HAS_ASYNC_LDS
#if __has_builtin(__builtin_amdgcn_s_wait_asynccnt)
  __builtin_amdgcn_s_wait_asynccnt(0);
#else
  asm volatile("s_wait_asynccnt 0" ::: "memory");
#endif
#endif
}

static __device__ __forceinline__ float softplus_f(float x) {
  return x > 0.f ? x + log1pf(__expf(-x)) : log1pf(__expf(x));
}

// Load 16 bf16 lane-elements as two 16-byte LDS chunks (A/B-matrix lane layout).
static __device__ __forceinline__ v16bf ld16_pair(const __bf16* p0, const __bf16* p1) {
  union { uint4 u[2]; v16bf v; } t;
  t.u[0] = *(const uint4*)p0;
  t.u[1] = *(const uint4*)p1;
  return t.v;
}

// ---------------------------------------------------------------------------
// Kernel 1: projection GEMM  P = X @ W^T  (f32 WMMA 16x16x4), fused epilogue.
//   MODE 0: Aq[n,h,l,0:64]=softplus(p)*cos(l*pw+pb), [64:128]=...*sin(...)
//   MODE 1: Bk[n,h,l,*]    =coeff*softplus(p)*{cos,sin}(l*pw)
//   MODE 2: Vb[n,h,l,0:64] =p
// Block: 256 threads = 8 waves; tile 32 rows x 64 cols (one head column block).
// ---------------------------------------------------------------------------
template <int MODE>
__global__ __launch_bounds__(256, 2) void fka_proj_kernel(
    const float* __restrict__ X, const float* __restrict__ W,
    const float* __restrict__ pw, const float* __restrict__ pb,
    const float* __restrict__ coeff, __bf16* __restrict__ outp)
{
  __shared__ __align__(16) float Xs[32 * 64];
  __shared__ __align__(16) float Ws[64 * 64];

  const int t    = threadIdx.x;
  const int lane = t & 31, wave = t >> 5;
  const int half = lane >> 4, ln = lane & 15;
  const int mq   = wave & 1;    // row sub-tile (0..1)
  const int nc   = wave >> 1;   // col sub-tile (0..3)
  const int row0 = blockIdx.x * 32;          // row in flattened (N*L)
  const int h    = blockIdx.y;               // head
  const int col0 = h * D_HEAD;               // column in D

  v8f acc = {0.f,0.f,0.f,0.f,0.f,0.f,0.f,0.f};

  for (int k0 = 0; k0 < D_MODEL; k0 += 64) {
    __syncthreads();
    // X tile: 32 x 64 f32, async direct to LDS (2 x b128 per thread).
    for (int i = t; i < 32 * 16; i += 256) {
      int r = i >> 4, c = (i & 15) << 2;
      copy16_g2l(&X[(size_t)(row0 + r) * D_MODEL + k0 + c], &Xs[r * 64 + c]);
    }
    // W tile: 64 x 64 f32 (4 x b128 per thread).
    for (int i = t; i < 64 * 16; i += 256) {
      int r = i >> 4, c = (i & 15) << 2;
      copy16_g2l(&W[(size_t)(col0 + r) * D_MODEL + k0 + c], &Ws[r * 64 + c]);
    }
    async_wait();
    __syncthreads();

    // A(16x4 f32): m=ln, k=e+2*half ; B(4x16): n=ln, k=e+2*half (Ws holds W[col][k])
    const float* xr = &Xs[(mq * 16 + ln) * 64 + 2 * half];
    const float* wr = &Ws[(nc * 16 + ln) * 64 + 2 * half];
#pragma unroll
    for (int kk = 0; kk < 64; kk += 4) {
      v2f a = *(const v2f*)(xr + kk);
      v2f b = *(const v2f*)(wr + kk);
      acc = __builtin_amdgcn_wmma_f32_16x16x4_f32(
          false, a, false, b, (short)0, acc, false, false);
    }
  }

  // Epilogue. C layout: m = r + 8*half, n = ln.
  const int dd = nc * 16 + ln;
  float pwv = 0.f, pbv = 0.f, cfv = 1.f;
  if (MODE != 2) {
    pwv = pw[h * D_HEAD + dd];
    if (MODE == 0) pbv = pb[h * D_HEAD + dd];
    else           cfv = coeff[h * D_HEAD + dd];
  }
#pragma unroll
  for (int r = 0; r < 8; r++) {
    const int gm = row0 + mq * 16 + r + 8 * half;
    const int nb = gm >> 11;                 // / L_SEQ
    const int l  = gm & (L_SEQ - 1);
    const float a = acc[r];
    if (MODE == 2) {
      outp[(((size_t)nb * H_N + h) * L_SEQ + l) * D_HEAD + dd] = (__bf16)a;
    } else {
      const float val = softplus_f(a);
      const float ph  = (MODE == 0) ? fmaf((float)l, pwv, pbv) : (float)l * pwv;
      float sn, cs;
      __sincosf(ph, &sn, &cs);
      __bf16* o = &outp[(((size_t)nb * H_N + h) * L_SEQ + l) * F_DIM];
      o[dd]          = (__bf16)(cfv * val * cs);
      o[D_HEAD + dd] = (__bf16)(cfv * val * sn);
    }
  }
}

// ---------------------------------------------------------------------------
// Kernel 2: causal attention with abs-sum normalization, fused single pass.
//   S = Aq_tile(64x128) @ Bk_tile(64x128)^T   (bf16 WMMA, K=128)
//   num += S @ V_tile (bf16 WMMA, K=64), den += rowsum(|S|)  -> out = num/den
// Block: 256 threads = 8 waves; one (n,h,qtile-of-64) per block.
// ---------------------------------------------------------------------------
__global__ __launch_bounds__(256, 2) void fka_attn_kernel(
    const __bf16* __restrict__ Aq, const __bf16* __restrict__ Bk,
    const __bf16* __restrict__ Vb, float* __restrict__ out)
{
  __shared__ __align__(16) __bf16 Aqs[64 * F_DIM];   // 16 KB
  __shared__ __align__(16) __bf16 Bks[64 * F_DIM];   // 16 KB
  __shared__ __align__(16) __bf16 Vts[64 * 64];      //  8 KB (transposed: [dd][k])
  __shared__ __align__(16) __bf16 Ss [64 * 64];      //  8 KB
  __shared__ float den[64];

  const int t    = threadIdx.x;
  const int lane = t & 31, wave = t >> 5;
  const int half = lane >> 4, ln = lane & 15;
  const int mq = wave >> 1;     // q sub-tile (0..3)
  const int pr = wave & 1;      // handles n-subtiles {2*pr, 2*pr+1}
  const int qt = blockIdx.x, h = blockIdx.y, nb = blockIdx.z;
  const int qstart = qt * 64;
  const size_t baseHL = ((size_t)nb * H_N + h) * L_SEQ;

  // Stage the (loop-invariant) Aq tile: 64 x 128 bf16, async direct to LDS.
  for (int i = t; i < 64 * 16; i += 256) {
    int r = i >> 4, c = (i & 15) << 3;
    copy16_g2l(&Aq[(baseHL + qstart + r) * F_DIM + c], &Aqs[r * F_DIM + c]);
  }
  if (t < 64) den[t] = 0.f;

  v8f num0 = {0.f,0.f,0.f,0.f,0.f,0.f,0.f,0.f};
  v8f num1 = {0.f,0.f,0.f,0.f,0.f,0.f,0.f,0.f};

  for (int kt = 0; kt <= qt; kt++) {
    const int kstart = kt * 64;
    __syncthreads();  // Aq/den ready (first iter); prior consumers done (later)

    // Stage Bk tile (64x128) async; V tile transposed (Vts[dd][k]) via registers.
    for (int i = t; i < 64 * 16; i += 256) {
      int r = i >> 4, c = (i & 15) << 3;
      copy16_g2l(&Bk[(baseHL + kstart + r) * F_DIM + c], &Bks[r * F_DIM + c]);
    }
    for (int i = t; i < 64 * 8; i += 256) {
      int r = i >> 3, c = (i & 7) << 3;
      union { uint4 u; __bf16 b[8]; } v;
      v.u = *(const uint4*)&Vb[(baseHL + kstart + r) * D_HEAD + c];
#pragma unroll
      for (int j = 0; j < 8; j++) Vts[(c + j) * 64 + r] = v.b[j];
    }
    // Prefetch next k-tile's Bk rows into cache while this tile computes.
    if (kt < qt) {
      const int r = t >> 2, c = (t & 3) << 5;
      __builtin_prefetch(&Bk[(baseHL + kstart + 64 + r) * F_DIM + c], 0, 1);
    }
    async_wait();
    __syncthreads();

    // ---- Stage 1: score tiles (K = 128 Fourier features) ----
#pragma unroll
    for (int p = 0; p < 2; p++) {
      const int nk = pr * 2 + p;
      v8f s = {0.f,0.f,0.f,0.f,0.f,0.f,0.f,0.f};
      const __bf16* arow = &Aqs[(mq * 16 + ln) * F_DIM];
      const __bf16* brow = &Bks[(nk * 16 + ln) * F_DIM + 16 * half];
#pragma unroll
      for (int kb = 0; kb < F_DIM; kb += 32) {
        v16bf a = ld16_pair(arow + kb + 16 * half, arow + kb + 16 * half + 8);
        v16bf b = ld16_pair(brow + kb, brow + kb + 8);
        s = __builtin_amdgcn_wmma_f32_16x16x32_bf16(
            false, a, false, b, (short)0, s, false, false);
      }
      // Causal mask + store S as bf16 (A-operand source for stage 2).
#pragma unroll
      for (int r = 0; r < 8; r++) {
        const int qg = qstart + mq * 16 + r + 8 * half;
        const int kg = kstart + nk * 16 + ln;
        const float sv = (kg > qg) ? 0.f : s[r];
        Ss[(mq * 16 + r + 8 * half) * 64 + nk * 16 + ln] = (__bf16)sv;
      }
    }
    __syncthreads();

    // ---- den += rowsum(|S|) (4 partial sums per row via LDS float atomics) ----
    {
      const int row = t & 63, quad = t >> 6;
      const __bf16* sr = &Ss[row * 64 + quad * 16];
      float dsum = 0.f;
#pragma unroll
      for (int j = 0; j < 16; j++) dsum += fabsf((float)sr[j]);
      atomicAdd(&den[row], dsum);
    }

    // ---- Stage 2: num += S(64x64) @ V(64x64) ----
    {
      const __bf16* arow = &Ss[(mq * 16 + ln) * 64];
#pragma unroll
      for (int p = 0; p < 2; p++) {
        const int nd = pr * 2 + p;
        const __bf16* brow = &Vts[(nd * 16 + ln) * 64 + 16 * half];
        v8f acc = p ? num1 : num0;
#pragma unroll
        for (int kb = 0; kb < 64; kb += 32) {
          v16bf a = ld16_pair(arow + kb + 16 * half, arow + kb + 16 * half + 8);
          v16bf b = ld16_pair(brow + kb, brow + kb + 8);
          acc = __builtin_amdgcn_wmma_f32_16x16x32_bf16(
              false, a, false, b, (short)0, acc, false, false);
        }
        if (p) num1 = acc; else num0 = acc;
      }
    }
  }
  __syncthreads();  // den finalized

  // Epilogue: out[n, q, h*64+dd] = num / den
#pragma unroll
  for (int p = 0; p < 2; p++) {
    const v8f acc = p ? num1 : num0;
    const int nd = pr * 2 + p;
#pragma unroll
    for (int r = 0; r < 8; r++) {
      const int ql = mq * 16 + r + 8 * half;
      const int dd = nd * 16 + ln;
      out[((size_t)nb * L_SEQ + qstart + ql) * D_MODEL + h * D_HEAD + dd] =
          acc[r] / den[ql];
    }
  }
}

// ---------------------------------------------------------------------------
extern "C" void kernel_launch(void* const* d_in, const int* in_sizes, int n_in,
                              void* d_out, int out_size, void* d_ws, size_t ws_size,
                              hipStream_t stream) {
  (void)in_sizes; (void)n_in; (void)out_size; (void)ws_size;
  const float* query = (const float*)d_in[0];
  const float* key   = (const float*)d_in[1];
  const float* Wq    = (const float*)d_in[2];
  const float* Wk    = (const float*)d_in[3];
  const float* Wv    = (const float*)d_in[4];
  const float* coeff = (const float*)d_in[5];
  const float* pw    = (const float*)d_in[6];  // position_weight (H,d,1)
  const float* pb    = (const float*)d_in[7];  // position_bias (H,d)

  // Workspace carve-out (bf16): Aq 8MB | Bk 8MB | Vb 4MB.
  const size_t AQ_BYTES = (size_t)N_B * H_N * L_SEQ * F_DIM * sizeof(__bf16);
  __bf16* Aq = (__bf16*)d_ws;
  __bf16* Bk = (__bf16*)((char*)d_ws + AQ_BYTES);
  __bf16* Vb = (__bf16*)((char*)d_ws + 2 * AQ_BYTES);

  dim3 blk(256);
  dim3 g1((N_B * L_SEQ) / 32, H_N);           // 128 x 8 blocks
  fka_proj_kernel<0><<<g1, blk, 0, stream>>>(query, Wq, pw, pb, coeff, Aq);
  fka_proj_kernel<1><<<g1, blk, 0, stream>>>(key,   Wk, pw, pb, coeff, Bk);
  fka_proj_kernel<2><<<g1, blk, 0, stream>>>(key,   Wv, pw, pb, coeff, Vb);

  dim3 g2(L_SEQ / 64, H_N, N_B);              // 32 x 8 x 2 blocks
  fka_attn_kernel<<<g2, blk, 0, stream>>>(Aq, Bk, Vb, (float*)d_out);
}
